// LambdaLayer_335007449673
// MI455X (gfx1250) — matrix operand order repaired and checked by reading, not compiled, not run on hardware
//
#include <hip/hip_runtime.h>
#include <hip/hip_bf16.h>

typedef __bf16 bf16_t;
typedef __attribute__((ext_vector_type(16))) __bf16 v16bf;
typedef __attribute__((ext_vector_type(8)))  __bf16 v8bf;
typedef __attribute__((ext_vector_type(4)))  __bf16 v4bf;
typedef __attribute__((ext_vector_type(8)))  float  v8f;
typedef __attribute__((ext_vector_type(4)))  unsigned int u32x4;
typedef __attribute__((ext_vector_type(8)))  int i32x8;
typedef __attribute__((ext_vector_type(4)))  int i32x4;

#define BATCH 8
#define DIM   256
#define MTOT  4096     // 64*64
#define EPSBN 1e-5f

// padded v_bn layout: [512 ch][86 rows][88 cols] bf16, image (iy,ix) at (iy+11, ix+11)
#define VROW  88
#define VPC   (86 * 88)          // 7568 elements per channel

// ---------------------------------------------------------------------------
// Kernel 0: prep — w_lambda f32 (16x529) -> bf16 padded [16][544];
//           zero lambda_c; zero padded-v buffer (margins must be 0).
// ---------------------------------------------------------------------------
__global__ void k_prep(const float* __restrict__ wl, bf16_t* __restrict__ wlb,
                       float* __restrict__ lam, unsigned int* __restrict__ vz)
{
    int idx0 = blockIdx.x * 256 + threadIdx.x;
    int stride = gridDim.x * 256;
    for (int idx = idx0; idx < 16 * 544; idx += stride) {
        int k = idx / 544, t = idx % 544;
        wlb[idx] = (t < 529) ? (bf16_t)wl[k * 529 + t] : (bf16_t)0.f;
    }
    for (int idx = idx0; idx < BATCH * 16 * 64; idx += stride) lam[idx] = 0.f;
    const int vwords = (512 * VPC) / 2;   // bf16 pairs as u32
    for (int idx = idx0; idx < vwords; idx += stride) vz[idx] = 0u;
}

// ---------------------------------------------------------------------------
// Kernel 1: qkv = W(144x256) @ x(256x4096) per batch, WMMA bf16, fused BN.
// Grid: B*32 blocks (m tiles of 128), 256 threads (8 waves, one 16-col subtile
// each, all 9 row-tiles, K chunked 4x64 through LDS; B fragment hoisted).
// ---------------------------------------------------------------------------
__global__ __launch_bounds__(256) void k_qkv(
    const float* __restrict__ x,  const float* __restrict__ w,
    const float* __restrict__ qg, const float* __restrict__ qb,
    const float* __restrict__ qm, const float* __restrict__ qv,
    const float* __restrict__ vg, const float* __restrict__ vb,
    const float* __restrict__ vm, const float* __restrict__ vv,
    float* __restrict__ q_out, float* __restrict__ k_out,
    bf16_t* __restrict__ v_pad)
{
    __shared__ bf16_t xs[64][128];     // K-chunk x m-tile
    __shared__ bf16_t wst[144][64];    // out-ch x K-chunk

    const int tid  = threadIdx.x;
    const int b    = blockIdx.x >> 5;
    const int m0   = (blockIdx.x & 31) * 128;
    const int wave = tid >> 5, lane = tid & 31;
    const int n0   = wave * 16;
    const int hs   = (lane >= 16) ? 1 : 0;
    const int rA   = lane & 15;

    v8f acc[9] = {};

    for (int kc = 0; kc < 4; ++kc) {
        // stage X chunk (coalesced float4 loads, convert to bf16)
        #pragma unroll
        for (int i = 0; i < 8; ++i) {
            int idx4 = tid + 256 * i;           // 2048 float4 total
            int row = idx4 >> 5, c4 = idx4 & 31;
            const float4 f = *(const float4*)(x +
                (size_t)(b * DIM + kc * 64 + row) * MTOT + m0 + c4 * 4);
            v4bf pk;
            pk[0] = (bf16_t)f.x; pk[1] = (bf16_t)f.y;
            pk[2] = (bf16_t)f.z; pk[3] = (bf16_t)f.w;
            *(v4bf*)&xs[row][c4 * 4] = pk;
        }
        // stage W chunk
        #pragma unroll
        for (int i = 0; i < 36; ++i) {
            int idx = tid + 256 * i;            // 144*64 total
            int row = idx >> 6, col = idx & 63;
            wst[row][col] = (bf16_t)w[row * DIM + kc * 64 + col];
        }
        __syncthreads();

        #pragma unroll
        for (int kk = 0; kk < 2; ++kk) {
            // B fragment: 32x16 bf16, lane = col, K striped in-lane (built once,
            // reused for all 9 row-tiles)
            const int kb  = kk * 32 + (hs ? 8 : 0);
            const int col = n0 + rA;
            v16bf bfr;
            #pragma unroll
            for (int i = 0; i < 16; ++i) bfr[i] = xs[kb + (i < 8 ? i : i + 8)][col];

            const int ab = kk * 32 + (hs ? 8 : 0);
            #pragma unroll
            for (int rt = 0; rt < 9; ++rt) {
                // A fragment: 16x32 bf16, lane = row, two contiguous 8-elt runs
                const v8bf alo = *(const v8bf*)&wst[rt * 16 + rA][ab];
                const v8bf ahi = *(const v8bf*)&wst[rt * 16 + rA][ab + 16];
                v16bf a;
                #pragma unroll
                for (int i = 0; i < 8; ++i) { a[i] = alo[i]; a[i + 8] = ahi[i]; }
                acc[rt] = __builtin_amdgcn_wmma_f32_16x16x32_bf16(
                    false, a, false, bfr, (short)0, acc[rt], false, false);
            }
        }
        __syncthreads();
    }

    // epilogue: split into q (BN), k (raw), v (BN -> bf16 into padded layout)
    const int m = m0 + n0 + rA;
    const int iy = m >> 6, ix = m & 63;
    #pragma unroll
    for (int rt = 0; rt < 9; ++rt) {
        #pragma unroll
        for (int j = 0; j < 8; ++j) {
            int o = rt * 16 + j + (hs ? 8 : 0);
            float val = acc[rt][j];
            if (o < 64) {
                float s = qg[o] * rsqrtf(qv[o] + EPSBN);
                q_out[(size_t)(b * 64 + o) * MTOT + m] = val * s + (qb[o] - qm[o] * s);
            } else if (o < 80) {
                k_out[(size_t)(b * 16 + (o - 64)) * MTOT + m] = val;
            } else {
                int c = o - 80;
                float s = vg[c] * rsqrtf(vv[c] + EPSBN);
                v_pad[(size_t)(b * 64 + c) * VPC + (iy + 11) * VROW + ix + 11] =
                    (bf16_t)(val * s + (vb[c] - vm[c] * s));
            }
        }
    }
}

// ---------------------------------------------------------------------------
// Kernel 2: row softmax over k logits. Grid: B*16 rows, 256 threads.
// ---------------------------------------------------------------------------
__global__ __launch_bounds__(256) void k_softmax(float* __restrict__ kbuf)
{
    __shared__ float red[256];
    const int tid = threadIdx.x;
    float* row = kbuf + (size_t)blockIdx.x * MTOT;
    float vals[16];
    float mx = -3.4e38f;
    #pragma unroll
    for (int i = 0; i < 16; ++i) { vals[i] = row[tid + 256 * i]; mx = fmaxf(mx, vals[i]); }
    red[tid] = mx; __syncthreads();
    for (int s = 128; s > 0; s >>= 1) {
        if (tid < s) red[tid] = fmaxf(red[tid], red[tid + s]);
        __syncthreads();
    }
    mx = red[0]; __syncthreads();
    float sum = 0.f;
    #pragma unroll
    for (int i = 0; i < 16; ++i) { vals[i] = __expf(vals[i] - mx); sum += vals[i]; }
    red[tid] = sum; __syncthreads();
    for (int s = 128; s > 0; s >>= 1) {
        if (tid < s) red[tid] += red[tid + s];
        __syncthreads();
    }
    float inv = 1.f / red[0];
    #pragma unroll
    for (int i = 0; i < 16; ++i) row[tid + 256 * i] = vals[i] * inv;
}

// ---------------------------------------------------------------------------
// Kernel 3: lambda_c[b,k,v] = sum_m ksoft[b,k,m] * v_bn[b,v,m].
// Grid: B * 16 m-chunks of 256; LDS-staged; atomicAdd f32 accumulation.
// ---------------------------------------------------------------------------
__global__ __launch_bounds__(256) void k_lambda_c(
    const float* __restrict__ ksoft, const bf16_t* __restrict__ v_pad,
    float* __restrict__ lam)
{
    __shared__ float  ks[16 * 256];
    __shared__ bf16_t vs[64 * 256];
    const int tid = threadIdx.x;
    const int b  = blockIdx.x >> 4;
    const int m0 = (blockIdx.x & 15) * 256;
    #pragma unroll
    for (int i = 0; i < 16; ++i) {
        int idx = tid + 256 * i;
        ks[idx] = ksoft[(size_t)(b * 16 + (idx >> 8)) * MTOT + m0 + (idx & 255)];
    }
    #pragma unroll
    for (int i = 0; i < 64; ++i) {
        int idx = tid + 256 * i;
        int v = idx >> 8, m = m0 + (idx & 255);
        vs[idx] = v_pad[(size_t)(b * 64 + v) * VPC + ((m >> 6) + 11) * VROW + (m & 63) + 11];
    }
    __syncthreads();
    #pragma unroll
    for (int i = 0; i < 4; ++i) {
        int p = tid + 256 * i;
        int k = p >> 6, v = p & 63;
        float s = 0.f;
        for (int j = 0; j < 256; ++j) s += ks[k * 256 + j] * (float)vs[v * 256 + j];
        atomicAdd(&lam[(size_t)b * 1024 + k * 64 + v], s);
    }
}

// ---------------------------------------------------------------------------
// Kernel 4: fused 23x23 conv (implicit-im2col WMMA bf16) + yc + yp combine.
// Grid: B * 64 rows * 8 v-chunks; 256 threads; dynamic LDS (~103 KB).
// Halo tile + weight tile staged by the Tensor Data Mover (2 descriptors),
// synchronized with s_wait_tensorcnt. Wave w owns v-channel w of the chunk.
// ---------------------------------------------------------------------------
__global__ __launch_bounds__(256) void k_conv_combine(
    const float* __restrict__ q_ws, const bf16_t* __restrict__ v_pad,
    const bf16_t* __restrict__ wlam, const float* __restrict__ lam_c,
    const float* __restrict__ blam, float* __restrict__ out)
{
    extern __shared__ unsigned char smem_raw[];
    bf16_t* vin = (bf16_t*)smem_raw;          // [8][23][88] halo tile of v_bn
    bf16_t* wt  = vin + 8 * 23 * 88;          // [16][544]   conv weights
    float*  P   = (float*)(wt + 16 * 544);    // [16][8][64] lambda_p tile
    float*  qs  = P + 16 * 8 * 64;            // [64][64]    q for this row
    float*  lcs = qs + 64 * 64;               // [16][64]    lambda_c
    float*  bls = lcs + 16 * 64;              // [16]        b_lambda

    const int tid = threadIdx.x;
    const int vc  = blockIdx.x & 7;
    const int y   = (blockIdx.x >> 3) & 63;
    const int b   = blockIdx.x >> 9;

#if __has_builtin(__builtin_amdgcn_tensor_load_to_lds)
    if (tid < 32) {   // wave 0 issues both TDM descriptors
        // --- vin: 3D tile [v=8][rows=23][cols=88] from padded v tensor ---
        unsigned lds_v = (unsigned)(size_t)vin;
        unsigned long long ga = (unsigned long long)(size_t)(v_pad +
            (size_t)(b * 64 + vc * 8) * VPC + (size_t)y * VROW);
        u32x4 g0 = { 1u, lds_v, (unsigned)ga,
                     (unsigned)(ga >> 32) | 0x80000000u };   // count=1, type=2
        i32x8 g1 = { 0x10000,                 // data_size=2B
                     (int)(88u << 16),        // tensor_dim0 = 88
                     (int)(23u << 16),        // tensor_dim1 = 23
                     (int)(88u << 16),        // tile_dim0   = 88
                     23 | (8 << 16),          // tile_dim1=23, tile_dim2=8
                     VROW,                    // tensor_dim0_stride = 88
                     (int)((unsigned)VPC << 16), // tensor_dim1_stride lo16
                     (int)((unsigned)VPC >> 16) };
        i32x4 g2 = { 8, 1, VPC, 0 };          // tensor_dim2=8, dim3=1, dim2_stride
        i32x4 g3 = { 0, 0, 0, 0 };
#if __clang_major__ >= 23
        i32x8 gz = {};
        __builtin_amdgcn_tensor_load_to_lds(g0, g1, g2, g3, gz, 0);
#else
        __builtin_amdgcn_tensor_load_to_lds(g0, g1, g2, g3, 0);
#endif
        // --- wt: 1D copy of 16x544 bf16 conv weights ---
        unsigned lds_w = (unsigned)(size_t)wt;
        unsigned long long gw = (unsigned long long)(size_t)wlam;
        u32x4 h0 = { 1u, lds_w, (unsigned)gw,
                     (unsigned)(gw >> 32) | 0x80000000u };
        i32x8 h1 = { 0x10000,
                     (int)(8704u << 16),      // tensor_dim0 = 8704
                     (int)(1u  << 16),        // tensor_dim1 = 1
                     (int)(8704u << 16),      // tile_dim0   = 8704
                     0,                        // tile_dim1/2 unused
                     8704, 0, 0 };
        i32x4 h2 = { 1, 1, 0, 0 };
        i32x4 h3 = { 0, 0, 0, 0 };
#if __clang_major__ >= 23
        __builtin_amdgcn_tensor_load_to_lds(h0, h1, h2, h3, gz, 0);
#else
        __builtin_amdgcn_tensor_load_to_lds(h0, h1, h2, h3, 0);
#endif
    }
#else
    // fallback: manual staging (no TDM builtin on this toolchain)
    for (int i = 0; i < 64; ++i) {
        int idx = tid + 256 * i;              // 8*23*88 = 16192
        if (idx < 16192) {
            int v = idx / 2024, rem = idx % 2024;
            int dy = rem / 88, c = rem % 88;
            vin[idx] = v_pad[(size_t)(b * 64 + vc * 8 + v) * VPC + (y + dy) * VROW + c];
        }
    }
    #pragma unroll
    for (int i = 0; i < 34; ++i) wt[tid + 256 * i] = wlam[tid + 256 * i];
#endif

    // small tiles staged manually (overlap with TDM)
    #pragma unroll
    for (int i = 0; i < 16; ++i) {
        int idx = tid + 256 * i;              // [ch=h*16+k][x]
        qs[idx] = q_ws[(size_t)(b * 64 + (idx >> 6)) * MTOT + y * 64 + (idx & 63)];
    }
    #pragma unroll
    for (int i = 0; i < 4; ++i) {
        int idx = tid + 256 * i;
        lcs[idx] = lam_c[(size_t)b * 1024 + idx];
    }
    if (tid < 16) bls[tid] = blam[tid];

#if __has_builtin(__builtin_amdgcn_s_wait_tensorcnt)
    __builtin_amdgcn_s_wait_tensorcnt(0);
#endif
    __syncthreads();

    // ---- conv as matmul: P_v(16x64) = W(16x544pad) @ im2col(544x64) ----
    const int v    = tid >> 5;                // wave id = local v channel
    const int lane = tid & 31;
    const int hs   = (lane >= 16) ? 1 : 0;
    const int rA   = lane & 15;

    for (int xt = 0; xt < 4; ++xt) {
        v8f acc = {};
        const int xcol = xt * 16 + rA;
        #pragma unroll
        for (int kk = 0; kk < 17; ++kk) {
            const int ab = kk * 32 + (hs ? 8 : 0);
            const v8bf alo = *(const v8bf*)&wt[rA * 544 + ab];
            const v8bf ahi = *(const v8bf*)&wt[rA * 544 + ab + 16];
            v16bf a;
            #pragma unroll
            for (int i = 0; i < 8; ++i) { a[i] = alo[i]; a[i + 8] = ahi[i]; }
            const int kb = kk * 32 + (hs ? 8 : 0);
            v16bf bfr;
            #pragma unroll
            for (int i = 0; i < 16; ++i) {
                int t = kb + (i < 8 ? i : i + 8); // compile-time constant after unroll
                if (t < 529) {
                    int dy = t / 23;
                    int dx = t - dy * 23;
                    bfr[i] = vin[v * 2024 + dy * 88 + xcol + dx];
                } else {
                    bfr[i] = (bf16_t)0.f;
                }
            }
            acc = __builtin_amdgcn_wmma_f32_16x16x32_bf16(
                false, a, false, bfr, (short)0, acc, false, false);
        }
        #pragma unroll
        for (int j = 0; j < 8; ++j) {
            int k = j + (hs ? 8 : 0);
            P[(k * 8 + v) * 64 + xcol] = acc[j];
        }
    }
    __syncthreads();

    // ---- combine: out[h,v,x] = sum_k q[h,k,x] * (P + lambda_c + b_lambda) ----
    #pragma unroll
    for (int i = 0; i < 8; ++i) {
        int idx = tid + 256 * i;              // 32 ch * 64 x
        int ch = idx >> 6, xx = idx & 63;
        int h = ch >> 3, vl = ch & 7;
        int vgl = vc * 8 + vl;
        float s = 0.f;
        #pragma unroll
        for (int k = 0; k < 16; ++k)
            s += qs[(h * 16 + k) * 64 + xx] *
                 (P[(k * 8 + vl) * 64 + xx] + lcs[k * 64 + vgl] + bls[k]);
        out[(size_t)(b * 256 + h * 64 + vgl) * MTOT + y * 64 + xx] = s;
    }
}

// ---------------------------------------------------------------------------
extern "C" void kernel_launch(void* const* d_in, const int* in_sizes, int n_in,
                              void* d_out, int out_size, void* d_ws, size_t ws_size,
                              hipStream_t stream)
{
    (void)in_sizes; (void)n_in; (void)out_size; (void)ws_size;
    const float* x  = (const float*)d_in[0];
    const float* w  = (const float*)d_in[1];
    const float* qg = (const float*)d_in[2];
    const float* qb = (const float*)d_in[3];
    const float* qm = (const float*)d_in[4];
    const float* qv = (const float*)d_in[5];
    const float* vg = (const float*)d_in[6];
    const float* vb = (const float*)d_in[7];
    const float* vm = (const float*)d_in[8];
    const float* vv = (const float*)d_in[9];
    const float* wl = (const float*)d_in[10];
    const float* bl = (const float*)d_in[11];
    float* out = (float*)d_out;

    unsigned char* ws = (unsigned char*)d_ws;
    float*  q_ws  = (float*)(ws);                    // 8*64*4096 f32   =  8 MB
    float*  k_ws  = (float*)(ws + 8388608);          // 8*16*4096 f32   =  2 MB
    bf16_t* v_ws  = (bf16_t*)(ws + 10485760);        // 512*7568 bf16   ~ 7.4 MB (padded)
    bf16_t* wl_ws = (bf16_t*)(ws + 18235392);        // 16*544 bf16
    float*  lc_ws = (float*)(ws + 18252800);         // 8*16*64 f32

    k_prep<<<512, 256, 0, stream>>>(wl, wl_ws, lc_ws, (unsigned int*)v_ws);
    k_qkv<<<BATCH * 32, 256, 0, stream>>>(x, w, qg, qb, qm, qv, vg, vb, vm, vv,
                                          q_ws, k_ws, v_ws);
    k_softmax<<<BATCH * 16, 256, 0, stream>>>(k_ws);
    k_lambda_c<<<BATCH * 16, 256, 0, stream>>>(k_ws, v_ws, lc_ws);

    size_t smem4 = (size_t)(8 * 23 * 88 + 16 * 544) * sizeof(bf16_t) +
                   (size_t)(16 * 8 * 64 + 64 * 64 + 16 * 64 + 16) * sizeof(float);
    k_conv_combine<<<BATCH * 64 * 8, 256, smem4, stream>>>(
        q_ws, v_ws, wl_ws, lc_ws, bl, out);
}
